// GaussianSplatter_13554916786177
// MI455X (gfx1250) — matrix operand order repaired and checked by reading, not compiled, not run on hardware
//
#include <hip/hip_runtime.h>
#include <hip/hip_bf16.h>

// ---------------------------------------------------------------------------
// GaussianSplatter on MI455X (gfx1250, wave32, WMMA)
//
// Shapes: inp[4][64][192][192] f32, sigma_x/y[2304], opacity[2304][1]
// ROW=COL=48, K=21, N=2304 points, BL=64 patches, C=64, HW=2304.
//
// Core: out[bl][c][hw] = clip( sum_p A[bl][c][p] * kt[p][hw] , 0, 1 )
//   A  = patch colors * opacity       (f16, [64][64][2304])
//   kt = translated separable Gaussian = S1[p][y]*S2[p][x]
//   Bm = kt transposed                 (f16, [hw=2304][p=2304])
// GEMM via v_wmma_f32_16x16x32_f16, fp32 accumulate.
// ---------------------------------------------------------------------------

typedef __attribute__((ext_vector_type(16))) _Float16 v16h;
typedef __attribute__((ext_vector_type(8)))  _Float16 v8h;
typedef __attribute__((ext_vector_type(8)))  float    v8f;

#define NPTS   2304      // ROW*COL
#define HWTOT  2304      // 48*48 patch pixels
#define CDIM   64
#define BLTOT  64        // B*L = 4*16

// ---- workspace layout (bytes) --------------------------------------------
#define WS_S1T  ((size_t)0)              // 48*2304 f32 = 442368 B   S1t[y][p]
#define WS_S2T  ((size_t)442368)         // 48*2304 f32              S2t[x][p]
#define WS_BM   ((size_t)884736)         // 2304*2304 f16 = 10616832 Bm[hw][p]
#define WS_AM   ((size_t)11501568)       // 64*64*2304 f16 = 18874368 Am[bl][c][p]
// total: 30375936 bytes

// 1-D Gaussian sample of the padded kernel: nonzero for t in [13,34)
__device__ __forceinline__ float gk(int t, float sigma) {
  if (t < 13 || t >= 34) return 0.0f;
  float ax = 0.5f * (float)(t - 13) - 5.0f;         // linspace(-5,5,21)
  return __expf(-0.5f * ax * ax / (sigma * sigma));
}

// K1: per point p and coordinate u (0..47) compute the lerped 1-D factors.
// S1 uses sigma_x + shift from i (row), S2 uses sigma_y + shift from j (col)
// (matches the reference's axis quirk). Stored transposed: S[u][p].
__global__ __launch_bounds__(256)
void gs_k1_lerp(const float* __restrict__ sigx, const float* __restrict__ sigy,
                float* __restrict__ S1t, float* __restrict__ S2t) {
  int gid = blockIdx.x * 256 + threadIdx.x;          // 48*2304 threads
  int p = gid % NPTS;
  int u = gid / NPTS;                                // 0..47
  int i = p / 48, j = p % 48;
  float sx = sigx[p], sy = sigy[p];

  // shift = 23.5 - (47/48)*idx  (from tx=1-2j/48, ty=1-2i/48, *(w-1)/2)
  float shy = 23.5f - (47.0f / 48.0f) * (float)i;
  float Sy  = floorf(shy);
  float fy  = shy - Sy;
  int   t0  = u + (int)Sy;
  S1t[gid] = (1.0f - fy) * gk(t0, sx) + fy * gk(t0 + 1, sx);

  float shx = 23.5f - (47.0f / 48.0f) * (float)j;
  float Sx  = floorf(shx);
  float fx  = shx - Sx;
  int   t1  = u + (int)Sx;
  S2t[gid] = (1.0f - fx) * gk(t1, sy) + fx * gk(t1 + 1, sy);
}

// K2: Bm[hw][p] = S1[p][y] * S2[p][x]   (f16, K-contiguous for B fragments)
__global__ __launch_bounds__(256)
void gs_k2_buildB(const float* __restrict__ S1t, const float* __restrict__ S2t,
                  _Float16* __restrict__ Bm) {
  int p  = blockIdx.x * 256 + threadIdx.x;           // gridDim.x = 9
  int hw = blockIdx.y;                               // gridDim.y = 2304
  int y = hw / 48, x = hw % 48;
  float v = S1t[y * NPTS + p] * S2t[x * NPTS + p];
  Bm[(size_t)hw * NPTS + p] = (_Float16)v;
}

// K3: Am[bl][c][p] = unfolded colors * opacity  (f16, K-contiguous rows)
__global__ __launch_bounds__(256)
void gs_k3_buildA(const float* __restrict__ inp, const float* __restrict__ opac,
                  _Float16* __restrict__ Am) {
  int idx = blockIdx.x * 256 + threadIdx.x;          // 64*64*2304 threads
  int p  = idx % NPTS;
  int c  = (idx / NPTS) % CDIM;
  int bl = idx / (NPTS * CDIM);
  int b = bl >> 4, patch = bl & 15, br = patch >> 2, bc = patch & 3;
  int r = p / 48, cl = p % 48;
  float v = inp[(((size_t)(b * 64 + c) * 192) + br * 48 + r) * 192 + bc * 48 + cl];
  Am[idx] = (_Float16)(v * opac[p]);
}

__device__ __forceinline__ v16h cat8(v8h lo, v8h hi) {
  return __builtin_shufflevector(lo, hi, 0,1,2,3,4,5,6,7,8,9,10,11,12,13,14,15);
}

// K4: one wave per 32x32 output macro-tile (2x2 WMMA tiles), K-loop over 2304.
// grid = (72 Ngroups, 2 Mgroups, 64 batches), block = 32 (one wave).
__global__ __launch_bounds__(32)
void gs_k4_gemm(const _Float16* __restrict__ Am, const _Float16* __restrict__ Bm,
                float* __restrict__ out) {
  const int lane = threadIdx.x;                      // 0..31
  const int hw0  = blockIdx.x * 32;                  // N base
  const int m0   = blockIdx.y * 32;                  // M base (channel)
  const int bl   = blockIdx.z;                       // patch-batch

  const int n  = lane & 15;
  const int kg = lane >> 4;                          // 0/1 half-wave group

  const _Float16* Ab = Am + (size_t)bl * CDIM * NPTS;
  // A fragment lane bases: row M = m0 + n, 16 contiguous K at kg*16
  const _Float16* a0p = Ab + (size_t)(m0 + n) * NPTS + kg * 16;
  const _Float16* a1p = a0p + (size_t)16 * NPTS;
  // B fragment lane bases: col N = hw0 + n, 16 contiguous K at kg*16
  const _Float16* b0p = Bm + (size_t)(hw0 + n) * NPTS + kg * 16;
  const _Float16* b1p = b0p + (size_t)16 * NPTS;

  v8f c00 = {}; v8f c01 = {}; v8f c10 = {}; v8f c11 = {};

#pragma unroll 4
  for (int k = 0; k < NPTS; k += 32) {
    v16h a0 = cat8(*(const v8h*)(a0p + k), *(const v8h*)(a0p + k + 8));
    v16h a1 = cat8(*(const v8h*)(a1p + k), *(const v8h*)(a1p + k + 8));
    v16h b0 = *(const v16h*)(b0p + k);               // 32B aligned
    v16h b1 = *(const v16h*)(b1p + k);
    c00 = __builtin_amdgcn_wmma_f32_16x16x32_f16(false, a0, false, b0, (short)0, c00, false, false);
    c01 = __builtin_amdgcn_wmma_f32_16x16x32_f16(false, a0, false, b1, (short)0, c01, false, false);
    c10 = __builtin_amdgcn_wmma_f32_16x16x32_f16(false, a1, false, b0, (short)0, c10, false, false);
    c11 = __builtin_amdgcn_wmma_f32_16x16x32_f16(false, a1, false, b1, (short)0, c11, false, false);
  }

  // store with clip + fold-back to [4][64][192][192]
  const int b = bl >> 4, patch = bl & 15, br = patch >> 2, bc = patch & 3;
#pragma unroll
  for (int mt = 0; mt < 2; ++mt) {
#pragma unroll
    for (int nt = 0; nt < 2; ++nt) {
      v8f acc = (mt == 0) ? (nt == 0 ? c00 : c01) : (nt == 0 ? c10 : c11);
      int hw = hw0 + nt * 16 + n;
      int y = hw / 48, x = hw % 48;
#pragma unroll
      for (int r = 0; r < 8; ++r) {
        int c = m0 + mt * 16 + kg * 8 + r;           // D layout: M = 8*kg + r
        float v = acc[r];
        v = fminf(fmaxf(v, 0.0f), 1.0f);
        out[(((size_t)(b * 64 + c) * 192) + br * 48 + y) * 192 + bc * 48 + x] = v;
      }
    }
  }
}

extern "C" void kernel_launch(void* const* d_in, const int* in_sizes, int n_in,
                              void* d_out, int out_size, void* d_ws, size_t ws_size,
                              hipStream_t stream) {
  const float* inp  = (const float*)d_in[0];
  const float* sigx = (const float*)d_in[1];
  const float* sigy = (const float*)d_in[2];
  const float* opac = (const float*)d_in[3];
  float* out = (float*)d_out;

  char* ws = (char*)d_ws;
  float*    S1t = (float*)(ws + WS_S1T);
  float*    S2t = (float*)(ws + WS_S2T);
  _Float16* Bm  = (_Float16*)(ws + WS_BM);
  _Float16* Am  = (_Float16*)(ws + WS_AM);

  // K1: 48*2304 = 110592 threads
  gs_k1_lerp<<<432, 256, 0, stream>>>(sigx, sigy, S1t, S2t);
  // K2: 2304x2304 outer products
  gs_k2_buildB<<<dim3(9, 2304), 256, 0, stream>>>(S1t, S2t, Bm);
  // K3: 64*64*2304 = 9437184 elements
  gs_k3_buildA<<<36864, 256, 0, stream>>>(inp, opac, Am);
  // K4: WMMA GEMM, 72 N-groups x 2 M-groups x 64 batches, 1 wave each
  gs_k4_gemm<<<dim3(72, 2, 64), 32, 0, stream>>>(Am, Bm, out);
}
